// RGB_Conv2d_79637283602878
// MI455X (gfx1250) — matrix operand order repaired and checked by reading, not compiled, not run on hardware
//
#include <hip/hip_runtime.h>
#include <math.h>

typedef __attribute__((ext_vector_type(16))) _Float16 v16h;
typedef __attribute__((ext_vector_type(8)))  float    v8f;

#define Bn 8
#define Cn 3
#define Hn 224
#define Wn 224
#define On 64
#define KHW 49        // 7x7
#define Ho 218
#define Wo 218
#define NPIX (Ho*Wo)            // 47524
#define TOTAL_P ((long)Bn*NPIX) // 380192 window rows
#define KAUG 128                // 49 (x.w) + 49 (x^2) + 1 (wsq) padded to 128
#define OUT_TOTAL ((long)Bn*On*NPIX)

// A-matrix element in augmented-K space for window row (b, y, x0), channel c.
__device__ __forceinline__ float aelem(const float* __restrict__ x,
                                       int b, int c, int y, int x0, int k) {
    if (k < KHW) {
        int ky = k / 7, kx = k - ky * 7;
        return x[((b * Cn + c) * Hn + y + ky) * Wn + (x0 + kx)];
    } else if (k < 2 * KHW) {
        int kk = k - KHW;
        int ky = kk / 7, kx = kk - ky * 7;
        float v = x[((b * Cn + c) * Hn + y + ky) * Wn + (x0 + kx)];
        return v * v;
    } else if (k == 2 * KHW) {
        return 1.0f;
    }
    return 0.0f;
}

// Kernel 1: dist[b,o,y,x] via augmented-K WMMA GEMM. 256 threads = 8 waves,
// each wave owns 16 window rows x all 64 outputs.
__global__ __launch_bounds__(256) void dist_kernel(const float* __restrict__ x,
                                                   const float* __restrict__ w,
                                                   float* __restrict__ dist) {
    // B staged transposed: sB[c][n][k] so each WMMA B-fragment is two 8-half
    // (16B) runs: k = (h<8 ? h : h+8) + 8*hi + 32*ks.
    __shared__ _Float16 sB[Cn][On][KAUG];
    __shared__ float    sWsq[On][Cn];

    const int tid = threadIdx.x;

    for (int t = tid; t < On * Cn; t += 256) {
        int n = t / Cn, c = t % Cn;
        float s = 0.f;
        for (int k = 0; k < KHW; ++k) {
            float ww = w[(n * Cn + c) * KHW + k];
            s += ww * ww;
        }
        sWsq[n][c] = s;
    }
    __syncthreads();
    for (int t = tid; t < Cn * On * KAUG; t += 256) {
        int c = t / (On * KAUG);
        int r = t % (On * KAUG);
        int n = r / KAUG;
        int k = r % KAUG;
        float val;
        if      (k < KHW)      val = -2.0f * w[(n * Cn + c) * KHW + k];
        else if (k < 2 * KHW)  val = 1.0f;
        else if (k == 2 * KHW) val = sWsq[n][c];
        else                   val = 0.0f;
        sB[c][n][k] = (_Float16)val;
    }
    __syncthreads();

    const int wave = tid >> 5;
    const int lane = tid & 31;
    const int hi   = lane >> 4;   // half-wave select
    const int mlo  = lane & 15;

    const long Pbase = ((long)blockIdx.x * 8 + wave) * 16;

    // A-load coordinates for row m = mlo (clamped for tail rows)
    long Pa  = Pbase + mlo;
    long Pac = (Pa < TOTAL_P) ? Pa : (TOTAL_P - 1);
    const int ab = (int)(Pac / NPIX);
    const int ap = (int)(Pac % NPIX);
    const int ay = ap / Wo;
    const int ax = ap % Wo;

    float distacc[4][8];
    #pragma unroll
    for (int t = 0; t < 4; ++t)
        #pragma unroll
        for (int e = 0; e < 8; ++e) distacc[t][e] = 0.0f;

    for (int c = 0; c < Cn; ++c) {
        v8f acc[4];
        #pragma unroll
        for (int t = 0; t < 4; ++t)
            acc[t] = (v8f){0.f, 0.f, 0.f, 0.f, 0.f, 0.f, 0.f, 0.f};

        #pragma unroll
        for (int ks = 0; ks < 4; ++ks) {
            // A fragment: half h -> k = (h<8 ? h : h+8) + 8*hi + 32*ks
            v16h af;
            #pragma unroll
            for (int h = 0; h < 16; ++h) {
                int k = ((h < 8) ? h : h + 8) + 8 * hi + 32 * ks;
                af[h] = (_Float16)aelem(x, ab, c, ay, ax, k);
            }
            #pragma unroll
            for (int t = 0; t < 4; ++t) {
                // B fragment: half h -> k = (h<8 ? h : h+8) + 8*hi + 32*ks, column n = mlo + 16*t
                v16h bf;
                #pragma unroll
                for (int h = 0; h < 16; ++h) {
                    int k = ((h < 8) ? h : h + 8) + 8 * hi + 32 * ks;
                    bf[h] = sB[c][t * 16 + mlo][k];
                }
                acc[t] = __builtin_amdgcn_wmma_f32_16x16x32_f16(
                    /*neg_a=*/false, af, /*neg_b=*/false, bf,
                    /*c_mod=*/(short)0, acc[t],
                    /*reuse_a=*/false, /*reuse_b=*/false);
            }
        }
        // acc == d2 for this channel; accumulate Euclidean distance
        #pragma unroll
        for (int t = 0; t < 4; ++t)
            #pragma unroll
            for (int e = 0; e < 8; ++e)
                distacc[t][e] += sqrtf(fmaxf(acc[t][e], 1e-12f));
    }

    // C/D layout: VGPR e -> row m = e + 8*hi; column n = mlo + 16*t
    #pragma unroll
    for (int e = 0; e < 8; ++e) {
        int  m = e + 8 * hi;
        long P = Pbase + m;
        if (P < TOTAL_P) {
            int b = (int)(P / NPIX);
            int p = (int)(P % NPIX);
            #pragma unroll
            for (int t = 0; t < 4; ++t) {
                int n = t * 16 + mlo;
                dist[((long)(b * On + n)) * NPIX + p] = distacc[t][e];
            }
        }
    }
}

// Kernel 2: per-pixel unbiased std over the (B*O)=512 samples.
__global__ __launch_bounds__(256) void std_kernel(const float* __restrict__ dist,
                                                  float* __restrict__ stdv) {
    int p = blockIdx.x * blockDim.x + threadIdx.x;
    if (p >= NPIX) return;
    float s = 0.f, ss = 0.f;
    for (int bo = 0; bo < Bn * On; ++bo) {
        float d = dist[(long)bo * NPIX + p];
        s += d;
        ss += d * d;
    }
    const float n    = (float)(Bn * On);
    float mean = s / n;
    float var  = (ss - n * mean * mean) / (n - 1.0f);
    stdv[p] = sqrtf(fmaxf(var, 0.0f));
}

// Kernel 3: in-place Gaussian RBF with per-pixel std broadcast.
__global__ __launch_bounds__(256) void gauss_kernel(float* __restrict__ out,
                                                    const float* __restrict__ stdv) {
    long i = (long)blockIdx.x * blockDim.x + threadIdx.x;
    if (i >= OUT_TOTAL) return;
    int   p = (int)(i % NPIX);
    float d = out[i];
    float s = stdv[p];
    out[i] = expf(-(d * d) / (2.0f * s * s));
}

extern "C" void kernel_launch(void* const* d_in, const int* in_sizes, int n_in,
                              void* d_out, int out_size, void* d_ws, size_t ws_size,
                              hipStream_t stream) {
    const float* x = (const float*)d_in[0];   // (8,3,224,224) f32
    const float* w = (const float*)d_in[1];   // (64,3,7,7)   f32
    float* out  = (float*)d_out;              // (8,64,218,218) f32; also holds dist
    float* stdv = (float*)d_ws;               // NPIX floats of scratch

    int blocks1 = (int)((TOTAL_P + 127) / 128);   // 128 rows per block (8 waves x 16)
    dist_kernel<<<blocks1, 256, 0, stream>>>(x, w, out);

    int blocks2 = (NPIX + 255) / 256;
    std_kernel<<<blocks2, 256, 0, stream>>>(out, stdv);

    int blocks3 = (int)((OUT_TOTAL + 255) / 256);
    gauss_kernel<<<blocks3, 256, 0, stream>>>(out, stdv);
}
